// TreeLstm_42812234007061
// MI455X (gfx1250) — matrix operand order repaired and checked by reading, not compiled, not run on hardware
//
#include <hip/hip_runtime.h>

// ---------------------------------------------------------------------------
// Problem constants (reference: B=128, L=256, V=50000, D=300, H=512)
// ---------------------------------------------------------------------------
#define BATCH  128
#define SEQ    256
#define DIM    300
#define DPAD   320          // D padded to a multiple of 32 for K-tiling
#define HID    512
#define GATES  2560         // 5*H
#define NTILES 160          // GATES / 16
#define OUTL   511          // 256+128+...+1

typedef __bf16        v16bf __attribute__((ext_vector_type(16)));
typedef float         v8f   __attribute__((ext_vector_type(8)));
typedef unsigned int  u32x4 __attribute__((ext_vector_type(4)));

union Frag { v16bf v; u32x4 q[2]; };   // 32 bytes: one WMMA A/B operand per lane

__device__ __forceinline__ unsigned short f2bf(float f) {
    unsigned u = __builtin_bit_cast(unsigned, f);
    u += 0x7FFFu + ((u >> 16) & 1u);          // round-to-nearest-even
    return (unsigned short)(u >> 16);
}

// Hardware transcendentals (v_exp_f32 / v_rcp_f32 / v_tanh_f32) with libm
// fallbacks; TRANS ops co-execute with XDL WMMA on CDNA5.
__device__ __forceinline__ float sigm(float x) {
#if __has_builtin(__builtin_amdgcn_exp2f) && __has_builtin(__builtin_amdgcn_rcpf)
    return __builtin_amdgcn_rcpf(1.0f + __builtin_amdgcn_exp2f(-1.442695041f * x));
#else
    return 1.0f / (1.0f + __expf(-x));
#endif
}
__device__ __forceinline__ float ftanh(float x) {
#if __has_builtin(__builtin_amdgcn_tanhf)
    return __builtin_amdgcn_tanhf(x);
#else
    return tanhf(x);
#endif
}

// ---------------------------------------------------------------------------
// Pack f32 weights (Kreal x 2560 row-major) into bf16 WMMA B-fragment tiles:
// tile(kt,nt) = 32(K) x 16(N), lane-major: each lane's 16 bf16 (32B) are
// contiguous -> two coalesced b128 loads per fragment. Lanes 0-15 hold
// K=kt*32+{0..7,16..23}, lanes 16-31 hold K=kt*32+{8..15,24..31} (ISA 16-bit B layout).
// ---------------------------------------------------------------------------
__global__ void pack_w(const float* __restrict__ U, unsigned short* __restrict__ P,
                       int Kreal, int total) {
    int idx = blockIdx.x * 256 + threadIdx.x;
    if (idx >= total) return;
    int tile = idx >> 9, within = idx & 511;
    int l = within >> 4, e = within & 15;
    int kt = tile / NTILES, nt = tile - kt * NTILES;
    int k = kt * 32 + ((e >> 3) << 4) + ((l >> 4) << 3) + (e & 7);
    int n = (nt << 4) + (l & 15);
    float v = (k < Kreal) ? U[(size_t)k * GATES + n] : 0.0f;
    P[idx] = f2bf(v);
}

// Gather embedding rows per token, convert to bf16, pad D 300 -> 320.
__global__ void embed_gather(const int* __restrict__ tok, const float* __restrict__ emb,
                             unsigned short* __restrict__ xbf) {
    int idx = blockIdx.x * 256 + threadIdx.x;       // exactly 32768*320 threads
    int m = idx / DPAD, d = idx - m * DPAD;
    float v = 0.0f;
    if (d < DIM) v = emb[(size_t)tok[m] * DIM + d];
    xbf[idx] = f2bf(v);
}

// ---------------------------------------------------------------------------
// Fully fused tree-LSTM level: GEMM + bias + cell, no gate scratch.
//   MODE 0 (leaf):     gates = x  @ Wx           (K = 320)
//   MODE 1 (internal): gates = hl @ Ul + hr @ Ur (K = 512, A rows = children)
// Block = 256 threads = 8 waves (2M x 4N) covering 32 rows x 64 H-columns.
// Each wave owns ALL FIVE gate tiles (i,fl,fr,o,u) for its 16 H-columns, so
// after the K loop it applies the LSTM cell entirely in registers.
// A tiles are staged to LDS once at entry via global_load_async_to_lds_b128
// (ASYNCcnt) + one barrier; the K loop is barrier-free so register-pipelined
// B fetches overlap WMMA issue with partial s_wait_loadcnt waits.
// LDS A layout is [unit][row]: element (row,unit) at unit*512 + row*16 bytes.
// A fragment read (fixed unit j0 = kt*4+hi) puts lanes 0-15 at row*16 ->
// banks 4r..4r+3 -> all 64 banks, conflict-free with NO swizzle, and the
// per-lane base is loop-invariant: kt*2048 folds into the ds offset field.
// ---------------------------------------------------------------------------
template<int MODE>
__global__ __launch_bounds__(256) void fused_level(
    const unsigned short* __restrict__ A,
    const unsigned short* __restrict__ BlP,
    const unsigned short* __restrict__ BrP,
    const float* __restrict__ bias,
    const float* __restrict__ cPrev,
    float* __restrict__ cOut,
    unsigned short* __restrict__ hbf,
    float* __restrict__ out,
    int nlShift, int outOff)
{
    constexpr int NCH  = (MODE == 1) ? 2 : 1;
    constexpr int KDIM = (MODE == 1) ? HID : DPAD;   // A row length (elements)
    constexpr int KT   = KDIM / 32;                  // K-steps
    constexpr int UPR  = KDIM / 8;                   // 16B units per A row
    constexpr int CH_EL = UPR * 256;                 // elements per child tile
    constexpr int CHUNKS = 32 * UPR * NCH;           // async b128 chunks per block
    constexpr int PER_T  = CHUNKS / 256;             // chunks per thread (5 or 16)

    __shared__ unsigned short sA[NCH][CH_EL];        // [unit][row] layout

    const int tid    = threadIdx.x;
    const int lane   = tid & 31;
    const int wave   = tid >> 5;
    const int waveM  = wave & 1;
    const int waveN  = wave >> 1;
    const int mb0    = blockIdx.y * 32;
    const int hc0    = blockIdx.x * 64 + waveN * 16; // wave's H-column base
    const int ntH    = hc0 >> 4;                     // 0..31
    const int lm15   = lane & 15;
    const int hi     = lane >> 4;

    // ---- stage entire A tile: one async b128 per chunk ----
    #pragma unroll
    for (int i = 0; i < PER_T; ++i) {
        const int c = tid + i * 256;
        int child, row, unit;
        if (MODE == 1) { child = c >> 11; row = (c >> 6) & 31; unit = c & 63; }
        else           { child = 0;       row = c / UPR;       unit = c % UPR; }
        const int m = mb0 + row;
        int src;
        if (MODE == 1) {
            int bb = m >> nlShift, nn = m & ((1 << nlShift) - 1);
            src = (bb << (nlShift + 1)) + 2 * nn + child;     // left/right child row
        } else {
            src = m;
        }
        unsigned lo = (unsigned)(size_t)&sA[child][unit * 256 + row * 8];
        unsigned long long ga =
            (unsigned long long)(size_t)(A + (size_t)src * KDIM + unit * 8);
        asm volatile("global_load_async_to_lds_b128 %0, %1, off" :: "v"(lo), "v"(ga) : "memory");
    }

    auto readA = [&](int child, int kt) -> Frag {    // 2x ds_load_b128, const base
        const unsigned short* base =
            &sA[(MODE == 1) ? child : 0][hi * 256 + (waveM * 16 + lm15) * 8];
        Frag f;
        f.q[0] = *(const u32x4*)(base + kt * 1024);        // unit kt*4+hi
        f.q[1] = *(const u32x4*)(base + kt * 1024 + 512);  // unit kt*4+hi+2
        return f;
    };
    // 5 gate fragments for this wave's 16 H-columns; gate tiles are 16384
    // elements apart (32 tiles * 512) in the packed weight layout.
    auto loadB5 = [&](const unsigned short* __restrict__ P, int kt, Frag b[5]) {
        const unsigned short* bp = P + (((size_t)kt * NTILES + ntH) << 9) + (lane << 4);
        #pragma unroll
        for (int gph = 0; gph < 5; ++gph) {
            b[gph].q[0] = *(const u32x4*)(bp + (gph << 14));
            b[gph].q[1] = *(const u32x4*)(bp + (gph << 14) + 8);
        }
    };

    v8f acc[5];                                      // gates i, fl, fr, o, u
    #pragma unroll
    for (int j = 0; j < 5; ++j)
        #pragma unroll
        for (int r = 0; r < 8; ++r) acc[j][r] = 0.0f;

    auto wmma5 = [&](const Frag& a, const Frag b[5]) {
        #pragma unroll
        for (int gph = 0; gph < 5; ++gph)
            acc[gph] = __builtin_amdgcn_wmma_f32_16x16x32_bf16(
                false, a.v, false, b[gph].v, (short)0, acc[gph], false, false);
    };

    Frag bX[5], bY[5];
    loadB5(BlP, 0, bX);

    // Single sync point: A tile resident, then a barrier-free K loop.
    asm volatile("s_wait_asynccnt 0x0" ::: "memory");
    __syncthreads();

    if (MODE == 1) {
        for (int kt = 0; kt < KT; ++kt) {
            loadB5(BrP, kt, bY);                            // Ur(kt) in flight...
            if (kt + 2 < KT)                                // near-scope prefetch 2 ahead
                __builtin_prefetch(BlP + (((size_t)(kt + 2) * NTILES + ntH) << 9), 0, 3);
            wmma5(readA(0, kt), bX);                        // ...while Ul WMMAs issue
            if (kt + 1 < KT) loadB5(BlP, kt + 1, bX);       // Ul(kt+1) in flight...
            wmma5(readA(1, kt), bY);                        // ...while Ur WMMAs issue
        }
    } else {
        for (int kt = 0; kt < KT; kt += 2) {
            if (kt + 1 < KT) loadB5(BlP, kt + 1, bY);
            wmma5(readA(0, kt), bX);
            if (kt + 2 < KT) loadB5(BlP, kt + 2, bX);
            if (kt + 1 < KT) wmma5(readA(0, kt + 1), bY);
        }
    }

    // ---- fused LSTM cell epilogue (all 5 gates in registers) ----
    // C/D layout: VGPR r -> row M = r (lanes 0-15) / 8+r (lanes 16-31); N = lane&15.
    const int col = hc0 + lm15;
    const float bi  = bias[col];
    const float bfl = bias[512  + col];
    const float bfr = bias[1024 + col];
    const float bo  = bias[1536 + col];
    const float bu  = bias[2048 + col];
    const int rowBase = mb0 + waveM * 16 + hi * 8;

    #pragma unroll
    for (int r = 0; r < 8; ++r) {
        const int m  = rowBase + r;
        const int bI = m >> nlShift;
        const int nI = m & ((1 << nlShift) - 1);

        const float gi  = acc[0][r] + bi;
        const float gfl = acc[1][r] + bfl;
        const float gfr = acc[2][r] + bfr;
        const float go  = acc[3][r] + bo;
        const float gu  = acc[4][r] + bu;

        float c = sigm(gi) * ftanh(gu);
        if (MODE == 1) {
            const size_t rL = ((size_t)(bI << (nlShift + 1)) + 2 * nI) * HID + col;
            c += sigm(gfl) * cPrev[rL] + sigm(gfr) * cPrev[rL + HID];
        }
        const float h = sigm(go) * ftanh(c);

        cOut[(size_t)m * HID + col] = c;
        hbf [(size_t)m * HID + col] = f2bf(h);
        out[((size_t)bI * OUTL + outOff + nI) * HID + col] = h;
    }
}

// ---------------------------------------------------------------------------
extern "C" void kernel_launch(void* const* d_in, const int* in_sizes, int n_in,
                              void* d_out, int out_size, void* d_ws, size_t ws_size,
                              hipStream_t stream) {
    (void)in_sizes; (void)n_in; (void)out_size; (void)ws_size;
    const int*   tok  = (const int*)  d_in[0];
    const float* emb  = (const float*)d_in[1];
    const float* Wx   = (const float*)d_in[2];
    const float* Ul   = (const float*)d_in[3];
    const float* Ur   = (const float*)d_in[4];
    const float* bias = (const float*)d_in[5];
    float*       out  = (float*)d_out;

    // Carve workspace (256 B aligned slices).
    size_t off = 0;
    auto carve = [&](size_t bytes) -> void* {
        void* p = (char*)d_ws + off;
        off += (bytes + 255) & ~(size_t)255;
        return p;
    };
    unsigned short* WxP = (unsigned short*)carve((size_t)10 * NTILES * 512 * 2); // 1.6 MB
    unsigned short* UlP = (unsigned short*)carve((size_t)16 * NTILES * 512 * 2); // 2.6 MB
    unsigned short* UrP = (unsigned short*)carve((size_t)16 * NTILES * 512 * 2); // 2.6 MB
    unsigned short* xbf = (unsigned short*)carve((size_t)BATCH * SEQ * DPAD * 2);// 21 MB
    unsigned short* hA  = (unsigned short*)carve((size_t)BATCH * SEQ * HID * 2); // 33.5 MB
    unsigned short* hB  = (unsigned short*)carve((size_t)BATCH * SEQ * HID * 2); // 33.5 MB
    float*          cA  = (float*)carve((size_t)BATCH * SEQ * HID * 4);          // 67 MB
    float*          cB  = (float*)carve((size_t)BATCH * SEQ * HID * 4);          // 67 MB

    // Weight packing + embedding gather.
    {
        int tWx = 10 * NTILES * 512, tU = 16 * NTILES * 512;
        pack_w<<<(tWx + 255) / 256, 256, 0, stream>>>(Wx, WxP, DIM, tWx);
        pack_w<<<(tU  + 255) / 256, 256, 0, stream>>>(Ul, UlP, HID, tU);
        pack_w<<<(tU  + 255) / 256, 256, 0, stream>>>(Ur, UrP, HID, tU);
        int tE = BATCH * SEQ * DPAD;                  // 10485760, multiple of 256
        embed_gather<<<tE / 256, 256, 0, stream>>>(tok, emb, xbf);
    }

    // Level 0 (leaves): gates = x @ Wx ; h,c with zero children. One launch.
    int outOff = 0;
    {
        dim3 grid(HID / 64, (BATCH * SEQ) / 32);      // (8, 1024)
        fused_level<0><<<grid, 256, 0, stream>>>(xbf, WxP, nullptr, bias,
                                                 nullptr, cA, hA, out, 8, 0);
        outOff = SEQ;
    }

    // Tree levels 1..8: Nl = 128, 64, ..., 1. One launch per level.
    unsigned short* hPrev = hA; float* cPrev = cA;
    unsigned short* hCur  = hB; float* cCur  = cB;
    for (int t = 1; t <= 8; ++t) {
        const int nlShift = 8 - t;
        const int Nl = 1 << nlShift;
        const int M = BATCH * Nl;
        dim3 grid(HID / 64, M / 32);
        fused_level<1><<<grid, 256, 0, stream>>>(hPrev, UlP, UrP, bias,
                                                 cPrev, cCur, hCur, out,
                                                 nlShift, outOff);
        outOff += Nl;
        unsigned short* th = hPrev; hPrev = hCur; hCur = th;
        float* tc = cPrev; cPrev = cCur; cCur = tc;
    }
}
